// TactileVQ_27401891348689
// MI455X (gfx1250) — matrix operand order repaired and checked by reading, not compiled, not run on hardware
//
#include <hip/hip_runtime.h>

typedef __attribute__((ext_vector_type(16))) __bf16 v16bf;
typedef __attribute__((ext_vector_type(8)))  __bf16 v8bf;
typedef __attribute__((ext_vector_type(8)))  float  v8f;
typedef unsigned int u32x4 __attribute__((ext_vector_type(4)));
typedef int          i32x4 __attribute__((ext_vector_type(4)));
typedef int          i32x8 __attribute__((ext_vector_type(8)));

#define D_DIM          256
#define K_CODES        1024
#define TILE_CODES     16
#define NUM_TILES      (K_CODES / TILE_CODES)     // 64
#define KSTEPS         (D_DIM / 32)               // 8 wmma k-steps
#define ROWS_PER_BLOCK 128                        // 8 waves * 16 rows
#define TILE_ELEMS     (TILE_CODES * D_DIM)       // 4096 bf16 = 8 KB

#if defined(__has_builtin)
#  if __has_builtin(__builtin_amdgcn_tensor_load_to_lds) && defined(__gfx1250__)
#    define VQ_USE_TDM 1
#  endif
#endif
#ifndef VQ_USE_TDM
#  define VQ_USE_TDM 0
#endif

// ---------------- prep: codebook fp32 -> bf16 + ||e||^2, zero loss ----------------
__global__ __launch_bounds__(256)
void vq_prep(const float* __restrict__ e, __bf16* __restrict__ eb,
             float* __restrict__ esq, float* __restrict__ loss) {
    const int row  = blockIdx.x * 8 + (threadIdx.x >> 5);   // one wave per code row
    const int lane = threadIdx.x & 31;
    const float* ep = e + (size_t)row * D_DIM + lane * 8;
    __bf16 tmp[8];
    float p = 0.0f;
    #pragma unroll
    for (int i = 0; i < 8; ++i) { float f = ep[i]; p = fmaf(f, f, p); tmp[i] = (__bf16)f; }
    *(v8bf*)&eb[(size_t)row * D_DIM + lane * 8] = *(v8bf*)tmp;
    #pragma unroll
    for (int m = 16; m >= 1; m >>= 1) p += __shfl_xor(p, m, 32);
    if (lane == 0) esq[row] = p;
    if (blockIdx.x == 0 && threadIdx.x == 0) *loss = 0.0f;
}

// ---------------- TDM issue: one 8 KB contiguous tile -> LDS ----------------
#if VQ_USE_TDM
typedef __attribute__((address_space(3))) __bf16 lds_bf16_t;

__device__ __forceinline__ void tdm_issue(const __bf16* eb, __bf16* dst, int t) {
    const unsigned lds_addr = (unsigned)(unsigned long long)(lds_bf16_t*)dst;
    const unsigned long long ga = (unsigned long long)(eb + (size_t)t * TILE_ELEMS);
    u32x4 g0;
    g0[0] = 1u;                                             // count=1, user descriptor
    g0[1] = lds_addr;                                       // lds_addr [63:32]
    g0[2] = (unsigned)ga;                                   // global_addr [95:64]
    g0[3] = (unsigned)((ga >> 32) & 0x01FFFFFFu) | (2u << 30); // addr[56:32] | type=2
    i32x8 g1;
    g1[0] = 0x10000;                  // workgroup_mask=0, data_size=1 (2B), no extras
    g1[1] = (int)(TILE_ELEMS << 16);  // tensor_dim0[15:0] in [31:16]
    g1[2] = 0x00010000;               // tensor_dim0[31:16]=0 ; tensor_dim1=1 (lo16)
    g1[3] = (int)(TILE_ELEMS << 16);  // tensor_dim1 hi=0 ; tile_dim0=4096
    g1[4] = 1;                        // tile_dim1=1, tile_dim2=0
    g1[5] = TILE_ELEMS;               // tensor_dim0_stride lo32
    g1[6] = 0;                        // stride hi, tensor_dim1_stride lo
    g1[7] = 0;
    i32x4 z4 = {0, 0, 0, 0};          // groups 2/3 unused (2D tensor)
#if __clang_major__ >= 23
    i32x8 z8 = {0, 0, 0, 0, 0, 0, 0, 0};
    __builtin_amdgcn_tensor_load_to_lds(g0, g1, z4, z4, z8, 0);
#else
    __builtin_amdgcn_tensor_load_to_lds(g0, g1, z4, z4, 0);
#endif
}
#endif

__device__ __forceinline__ void stage_tile(const __bf16* __restrict__ eb,
                                           __bf16* dst, int t, int tid) {
#if VQ_USE_TDM
    if ((tid >> 5) == 0) tdm_issue(eb, dst, t);             // wave 0 drives the TDM
#else
    // fallback: 256 threads x 32 B vector copy (tile is contiguous in memory)
    const v16bf* g = (const v16bf*)(eb + (size_t)t * TILE_ELEMS);
    ((v16bf*)dst)[tid] = g[tid];
    if (t + 1 < NUM_TILES)
        __builtin_prefetch(eb + (size_t)(t + 1) * TILE_ELEMS + tid * 16, 0, 0);
#endif
}

__device__ __forceinline__ void wait_tile(int tid) {
#if VQ_USE_TDM
    if ((tid >> 5) == 0) __builtin_amdgcn_s_wait_tensorcnt(0);
#endif
}

// ---------------- main: WMMA distances + argmin + gather + loss ----------------
__global__ __launch_bounds__(256)
void vq_main(const float* __restrict__ z, const float* __restrict__ e,
             const __bf16* __restrict__ eb, const float* __restrict__ esq,
             float* __restrict__ out, float* __restrict__ loss, float inv_count) {
    __shared__ __align__(64) __bf16 lds_e[2][TILE_ELEMS];   // double-buffered 8 KB tiles

    const int tid  = threadIdx.x;
    const int wave = tid >> 5;
    const int lane = tid & 31;
    const int half = lane >> 4;
    const int l16  = lane & 15;

    const int row0  = blockIdx.x * ROWS_PER_BLOCK + wave * 16;
    const int myrow = row0 + l16;                 // A-matrix row held by this lane

    // ---- load this wave's 16 z-rows as bf16 into WMMA 16-bit A layout ----
    v16bf a[KSTEPS];
    {
        const float* zr = z + (size_t)myrow * D_DIM;
        #pragma unroll
        for (int c = 0; c < KSTEPS; ++c) {
            const float* p0 = zr + 32 * c + 16 * half;       // k = 16h .. 16h+7
            const float* p1 = zr + 32 * c + 16 * half + 8;   // k = 16h+8 .. +7
            v16bf av;
            #pragma unroll
            for (int i = 0; i < 8; ++i) { av[i] = (__bf16)p0[i]; av[8 + i] = (__bf16)p1[i]; }
            a[c] = av;
        }
    }

    float best[8];
    int   bidx[8];
    #pragma unroll
    for (int r = 0; r < 8; ++r) { best[r] = 3.4e38f; bidx[r] = 0; }

    stage_tile(eb, &lds_e[0][0], 0, tid);                    // prologue: tile 0 -> buf 0

    for (int t = 0; t < NUM_TILES; ++t) {
        wait_tile(tid);                                      // tile t landed (TDM)
        __syncthreads();                                     // visible to all waves
        if (t + 1 < NUM_TILES)                               // overlap DMA of t+1 with compute
            stage_tile(eb, &lds_e[(t + 1) & 1][0], t + 1, tid);

        const __bf16* buf = &lds_e[t & 1][0];
        v8f acc = {};
        #pragma unroll
        for (int c = 0; c < KSTEPS; ++c) {
            // B layout: lane = code column (l16); v0..7 hold k = 16*half + [0..16)
            v16bf b = *(const v16bf*)&buf[l16 * D_DIM + 32 * c + 16 * half];
            acc = __builtin_amdgcn_wmma_f32_16x16x32_bf16(
                      false, a[c], false, b, (short)0, acc, false, false);
        }

        // dist' = ||e||^2 - 2 z.e   (||z||^2 is row-constant -> drops from argmin)
        const float eq  = esq[t * TILE_CODES + l16];
        const int   col = t * TILE_CODES + l16;
        #pragma unroll
        for (int r = 0; r < 8; ++r) {
            float dist = fmaf(-2.0f, acc[r], eq);
            if (dist < best[r]) { best[r] = dist; bidx[r] = col; }
        }
        __syncthreads();                                     // done reading buf[t&1]
    }

    // ---- argmin across the 16 lanes (columns) of each half, per row slot r ----
    #pragma unroll
    for (int r = 0; r < 8; ++r) {
        #pragma unroll
        for (int m = 1; m <= 8; m <<= 1) {
            float od = __shfl_xor(best[r], m, 16);
            int   oi = __shfl_xor(bidx[r], m, 16);
            if (od < best[r] || (od == best[r] && oi < bidx[r])) { best[r] = od; bidx[r] = oi; }
        }
    }

    // ---- gather winning fp32 code rows, write z_q, accumulate commitment loss ----
    float lsum = 0.0f;
    #pragma unroll
    for (int r = 0; r < 8; ++r) {
        const int row  = row0 + r + 8 * half;
        const int code = bidx[r];
        const float4* cq = (const float4*)(e   + (size_t)code * D_DIM) + l16 * 4;
        const float4* zz = (const float4*)(z   + (size_t)row  * D_DIM) + l16 * 4;
        float4*       oo = (float4*)      (out + (size_t)row  * D_DIM) + l16 * 4;
        #pragma unroll
        for (int q = 0; q < 4; ++q) {
            float4 cv = cq[q], zv = zz[q];
            oo[q] = cv;
            float dx = cv.x - zv.x, dy = cv.y - zv.y, dz = cv.z - zv.z, dw = cv.w - zv.w;
            lsum += dx * dx + dy * dy + dz * dz + dw * dw;
        }
    }
    #pragma unroll
    for (int m = 16; m >= 1; m >>= 1) lsum += __shfl_xor(lsum, m, 32);
    if (lane == 0) atomicAdd(loss, lsum * inv_count);
}

extern "C" void kernel_launch(void* const* d_in, const int* in_sizes, int n_in,
                              void* d_out, int out_size, void* d_ws, size_t ws_size,
                              hipStream_t stream) {
    (void)n_in; (void)out_size; (void)ws_size;
    const float* z   = (const float*)d_in[0];   // [B*T, D] fp32
    const float* e   = (const float*)d_in[1];   // [K, D]   fp32
    float*       out = (float*)d_out;           // z_q (N*D floats) ++ loss (1 float)
    const int    N   = in_sizes[0] / D_DIM;     // 65536
    float*       loss = out + (size_t)in_sizes[0];
    const float  inv_count = 1.0f / (float)in_sizes[0];

    __bf16* eb  = (__bf16*)d_ws;                              // 512 KB bf16 codebook
    float*  esq = (float*)((char*)d_ws + (size_t)K_CODES * D_DIM * sizeof(__bf16));

    vq_prep<<<dim3(K_CODES / 8), dim3(256), 0, stream>>>(e, eb, esq, loss);
    vq_main<<<dim3(N / ROWS_PER_BLOCK), dim3(256), 0, stream>>>(z, e, eb, esq, out,
                                                                loss, inv_count);
}
